// TransfomerBlock_3796751090378
// MI455X (gfx1250) — matrix operand order repaired and checked by reading, not compiled, not run on hardware
//
#include <hip/hip_runtime.h>
#include <hip/hip_bf16.h>
#include <math.h>

// ---------------- problem constants ----------------
#define B_   2
#define S_   2048
#define D_   1024
#define H_   16
#define HD_  64
#define DFF_ 4096
#define M_   (B_ * S_)      // 4096 token rows
#define EPSF 1e-5f

#define USE_ASYNC_COPY 1

#if defined(__has_builtin)
#if __has_builtin(__builtin_amdgcn_tensor_load_to_lds) && __has_builtin(__builtin_amdgcn_s_wait_tensorcnt)
#define HAVE_TDM 1
#endif
#endif
#ifndef HAVE_TDM
#define HAVE_TDM 0
#endif

typedef __bf16 bf16_t;
typedef __attribute__((ext_vector_type(16))) __bf16 v16bf;
typedef __attribute__((ext_vector_type(8)))  __bf16 v8bf;
typedef __attribute__((ext_vector_type(8)))  float  v8f;
typedef unsigned int u32x4 __attribute__((ext_vector_type(4)));
typedef int          i32x8 __attribute__((ext_vector_type(8)));
typedef int          i32x4 __attribute__((ext_vector_type(4)));

// ---------------- helpers ----------------
__device__ inline bf16_t f2bf(float f) {
    unsigned u = __float_as_uint(f);
    u += 0x7FFFu + ((u >> 16) & 1u);          // round-to-nearest-even
    unsigned short h = (unsigned short)(u >> 16);
    return __builtin_bit_cast(bf16_t, h);
}

// async 16B global -> LDS copy (CDNA5 ASYNCcnt path), with synchronous fallback
__device__ inline void stage_b128(bf16_t* lds_dst, const bf16_t* g) {
#if USE_ASYNC_COPY
    unsigned off = (unsigned)(size_t)lds_dst;              // LDS byte offset (addr[31:0])
    unsigned long long ga = (unsigned long long)(size_t)g;
    asm volatile("global_load_async_to_lds_b128 %0, %1, off"
                 :: "v"(off), "v"(ga) : "memory");
#else
    *(v8bf*)lds_dst = *(const v8bf*)g;
#endif
}
__device__ inline void wait_stage() {
#if USE_ASYNC_COPY
    asm volatile("s_wait_asynccnt 0x0" ::: "memory");
#endif
}

// A-fragment (16x32 bf16, contiguous K layout matching B):
// lane<16: row=lane, K=0..15; lane>=16: K=16..31.
__device__ inline v16bf load_a_bf16(const bf16_t* base, int ld, int lane) {
    int row = lane & 15;
    int kb  = (lane >> 4) << 4;               // 0 or 16
    const bf16_t* p = base + (size_t)row * ld + kb;
    v8bf lo = *(const v8bf*)p;                // K = kb .. kb+7
    v8bf hi = *(const v8bf*)(p + 8);          // K = kb+8 .. kb+15
    return __builtin_shufflevector(lo, hi, 0,1,2,3,4,5,6,7,8,9,10,11,12,13,14,15);
}

// A-fragment built from f32 source (converted on load)
__device__ inline v16bf load_a_f32(const float* base, int ld, int lane) {
    int row = lane & 15;
    int kb  = (lane >> 4) << 4;
    const float* p = base + (size_t)row * ld + kb;
    v16bf f;
#pragma unroll
    for (int i = 0; i < 8; ++i) { f[i] = f2bf(p[i]); f[8 + i] = f2bf(p[8 + i]); }
    return f;
}

// B-fragment (32x16 bf16): lane<16: col=lane, K=0..15; lane>=16: K=16..31 (contiguous 32B)
__device__ inline v16bf load_b_bf16(const bf16_t* base, int ld, int lane) {
    int col = lane & 15;
    int kb  = (lane >> 4) << 4;               // 0 or 16
    return *(const v16bf*)(base + (size_t)col * ld + kb);
}

__device__ inline v8f wmma_bf16(v16bf a, v16bf b, v8f c) {
    return __builtin_amdgcn_wmma_f32_16x16x32_bf16(false, a, false, b, (short)0, c, false, false);
}

__device__ inline float gelu_exact(float v) {
    return 0.5f * v * (1.0f + erff(v * 0.70710678118654752f));
}

// ---------------- elementwise / layout kernels ----------------
__global__ void k_cvt_bf16(const float* __restrict__ in, bf16_t* __restrict__ out, int n) {
    int i = blockIdx.x * blockDim.x + threadIdx.x;
    if (i < n) out[i] = f2bf(in[i]);
}

// W[Kr x Nc] f32  ->  Wt[Nc x Kr] bf16  (B-operand layout)
__global__ void k_transpose_cvt(const float* __restrict__ W, bf16_t* __restrict__ Wt,
                                int Kr, int Nc) {
    __shared__ float tile[32][33];
    int n0 = blockIdx.x * 32, k0 = blockIdx.y * 32;
    for (int i = threadIdx.y; i < 32; i += 8)
        tile[i][threadIdx.x] = W[(size_t)(k0 + i) * Nc + n0 + threadIdx.x];
    __syncthreads();
    for (int i = threadIdx.y; i < 32; i += 8)
        Wt[(size_t)(n0 + i) * Kr + k0 + threadIdx.x] = f2bf(tile[threadIdx.x][i]);
}

// V[b,s,h*64+d] bf16 -> Vt[(b*H+h)*64+d, s] bf16
__global__ void k_transpose_v(const bf16_t* __restrict__ V, bf16_t* __restrict__ Vt) {
    __shared__ bf16_t tile[32][33];
    int bh = blockIdx.z, b = bh / H_, h = bh % H_;
    int s0 = blockIdx.x * 32, d0 = blockIdx.y * 32;
    for (int i = threadIdx.y; i < 32; i += 8)
        tile[i][threadIdx.x] = V[((size_t)(b * S_) + s0 + i) * D_ + h * HD_ + d0 + threadIdx.x];
    __syncthreads();
    for (int i = threadIdx.y; i < 32; i += 8)
        Vt[((size_t)(bh * HD_) + d0 + i) * S_ + s0 + threadIdx.x] = tile[threadIdx.x][i];
}

// mask (S,S) int32 -> bitmask words (bit=1 where mask==1)
__global__ void k_maskbits(const int* __restrict__ mask, unsigned* __restrict__ bits) {
    int s = blockIdx.x, t0 = threadIdx.x * 32;   // blockDim.x == 64
    unsigned w = 0;
#pragma unroll 4
    for (int i = 0; i < 32; ++i)
        if (mask[(size_t)s * S_ + t0 + i] == 1) w |= (1u << i);
    bits[(size_t)s * (S_ / 32) + threadIdx.x] = w;
}

// ---------------- WMMA GEMM: C[M,N] = act(A[M,K] * Bt[N,K]^T + bias) ----------------
// 256 threads = 8 waves, 128x128 C tile per block; wave = 32(M) x 64(N) of it.
// Double-buffered LDS staged via async global->LDS copies.
__global__ __launch_bounds__(256) void k_gemm(
    const bf16_t* __restrict__ A, const bf16_t* __restrict__ Bt,
    const float* __restrict__ bias, float* __restrict__ outF, bf16_t* __restrict__ outB,
    int Mn, int Nn, int Kn, int act)
{
    __shared__ __align__(32) bf16_t As[2][128 * 40];   // row stride 40 elems (80B)
    __shared__ __align__(32) bf16_t Bs[2][128 * 48];   // row stride 48 elems (96B, 32B-aligned)
    const int tid = threadIdx.x, lane = tid & 31, w = tid >> 5;
    const int wm = w >> 1, wn = w & 1;
    const int m0 = blockIdx.y * 128, n0 = blockIdx.x * 128;

    // staging slots: this thread copies rows r and r+64, 16B column-chunk q
    const int r = tid >> 2, q = tid & 3;
    const bf16_t* gA0 = A  + (size_t)(m0 + r) * Kn + q * 8;
    const bf16_t* gB0 = Bt + (size_t)(n0 + r) * Kn + q * 8;
    const size_t rowskip = (size_t)64 * Kn;
    bf16_t* lA = &As[0][r * 40 + q * 8];
    bf16_t* lB = &Bs[0][r * 48 + q * 8];

    v8f acc[2][4];
#pragma unroll
    for (int mt = 0; mt < 2; ++mt)
#pragma unroll
        for (int nt = 0; nt < 4; ++nt)
#pragma unroll
            for (int j = 0; j < 8; ++j) acc[mt][nt][j] = 0.0f;

    // prologue: stage k-slice 0 into buffer 0
    stage_b128(lA,             gA0);
    stage_b128(lA + 64 * 40,   gA0 + rowskip);
    stage_b128(lB,             gB0);
    stage_b128(lB + 64 * 48,   gB0 + rowskip);

    for (int k0 = 0; k0 < Kn; k0 += 32) {
        const int buf = (k0 >> 5) & 1;
        wait_stage();            // own async copies for this k-slice done
        __syncthreads();         // everyone's copies visible
        if (k0 + 32 < Kn) {      // stage next k-slice into the other buffer
            const int nk = k0 + 32;
            bf16_t* nA = &As[buf ^ 1][r * 40 + q * 8];
            bf16_t* nB = &Bs[buf ^ 1][r * 48 + q * 8];
            stage_b128(nA,           gA0 + nk);
            stage_b128(nA + 64 * 40, gA0 + nk + rowskip);
            stage_b128(nB,           gB0 + nk);
            stage_b128(nB + 64 * 48, gB0 + nk + rowskip);
        }
        v16bf af[2], bfr[4];
#pragma unroll
        for (int mt = 0; mt < 2; ++mt)
            af[mt]  = load_a_bf16(&As[buf][(wm * 32 + mt * 16) * 40], 40, lane);
#pragma unroll
        for (int nt = 0; nt < 4; ++nt)
            bfr[nt] = load_b_bf16(&Bs[buf][(wn * 64 + nt * 16) * 48], 48, lane);
#pragma unroll
        for (int mt = 0; mt < 2; ++mt)
#pragma unroll
            for (int nt = 0; nt < 4; ++nt)
                acc[mt][nt] = wmma_bf16(af[mt], bfr[nt], acc[mt][nt]);
    }

#pragma unroll
    for (int mt = 0; mt < 2; ++mt) {
#pragma unroll
        for (int nt = 0; nt < 4; ++nt) {
            int col = n0 + wn * 64 + nt * 16 + (lane & 15);
            float bv = bias ? bias[col] : 0.0f;
            int rbase = m0 + wm * 32 + mt * 16 + ((lane >> 4) << 3);
#pragma unroll
            for (int j = 0; j < 8; ++j) {
                float v = acc[mt][nt][j] + bv;
                if (act) v = gelu_exact(v);
                size_t idx = (size_t)(rbase + j) * Nn + col;
                if (outF) outF[idx] = v;
                if (outB) outB[idx] = f2bf(v);
            }
        }
    }
}

// ---------------- flash attention: 64 q-rows x full S, kv chunks of 128 ----------------
__global__ __launch_bounds__(256) void k_attn(
    const bf16_t* __restrict__ Qb, const bf16_t* __restrict__ Kb,
    const bf16_t* __restrict__ Vt, const unsigned* __restrict__ mbits,
    bf16_t* __restrict__ Ob)
{
    __shared__ __align__(32) bf16_t Qs[64 * 72];
    __shared__ float  Sbuf[64 * 132];
    __shared__ float  m_s[64], l_s[64], a_s[64];
    __shared__ float  red[64 * 4];

    const int tid = threadIdx.x, lane = tid & 31, w = tid >> 5;
    const int q0 = blockIdx.x * 64;
    const int bh = blockIdx.y, b = bh / H_, h = bh % H_;
    const int qt = w & 3, hh = w >> 2;        // wave owns q-tile qt, t/hd half hh

#if HAVE_TDM
    // TDM: 64x64 bf16 tile, row stride D, LDS rows padded to 144B (pad 4 DWORDs / 32 DWORDs)
    if (w == 0) {
        unsigned long long ga =
            (unsigned long long)(size_t)(Qb + ((size_t)(b * S_) + q0) * D_ + h * HD_);
        u32x4 g0; i32x8 g1; i32x4 z4 = {0, 0, 0, 0};
        i32x8 z8 = {0, 0, 0, 0, 0, 0, 0, 0};
        g0[0] = 1u;                                   // count=1, user mode
        g0[1] = (unsigned)(size_t)Qs;                 // lds_addr (bytes)
        g0[2] = (unsigned)(ga & 0xFFFFFFFFull);       // global_addr[31:0]
        g0[3] = (unsigned)((ga >> 32) & 0x01FFFFFFull) | 0x80000000u; // addr[56:32] | type=2
        g1[0] = (int)((1u << 16) | (1u << 20) | (4u << 22) | (3u << 25)); // 2B elems, pad on, 32dw/4dw
        g1[1] = (int)(64u << 16);                     // tensor_dim0 = 64 (low 16 bits)
        g1[2] = (int)(64u << 16);                     // tensor_dim1 = 64
        g1[3] = (int)(64u << 16);                     // tile_dim0 = 64
        g1[4] = 64;                                   // tile_dim1 = 64
        g1[5] = (int)D_;                              // tensor_dim0_stride = 1024 elems
        g1[6] = 0; g1[7] = 0;
        __builtin_amdgcn_tensor_load_to_lds(g0, g1, z4, z4, z8, 0);
        __builtin_amdgcn_s_wait_tensorcnt(0);
    }
#else
    for (int c = tid; c < 512; c += 256) {    // Q block 64x64 bf16 -> LDS
        int rr = c >> 3, seg = c & 7;
        *(v8bf*)&Qs[rr * 72 + seg * 8] =
            *(const v8bf*)&Qb[((size_t)(b * S_) + q0 + rr) * D_ + h * HD_ + seg * 8];
    }
#endif
    if (tid < 64) { m_s[tid] = -3.0e38f; l_s[tid] = 0.0f; }
    v8f oacc[2];
#pragma unroll
    for (int e = 0; e < 2; ++e)
#pragma unroll
        for (int j = 0; j < 8; ++j) oacc[e][j] = 0.0f;
    __syncthreads();

    for (int t0 = 0; t0 < S_; t0 += 128) {
        // prefetch next kv chunk (K rows and Vt columns) while we compute this one
        if (t0 + 128 < S_) {
            __builtin_prefetch(Kb + ((size_t)(b * S_) + t0 + 128 + (w << 4)) * D_ + h * HD_, 0, 1);
            __builtin_prefetch(Vt + ((size_t)(bh * HD_) + (w << 3)) * S_ + t0 + 128, 0, 1);
        }
        // ---- scores: S = (Q K^T) / 8, masked; wave writes rows qt*16..+16, cols hh*64..+64
        v16bf qf[2];
#pragma unroll
        for (int kk = 0; kk < 2; ++kk)
            qf[kk] = load_a_bf16(&Qs[(qt * 16) * 72 + kk * 32], 72, lane);
#pragma unroll
        for (int u = 0; u < 4; ++u) {
            int tt = hh * 4 + u;
            v8f sacc;
#pragma unroll
            for (int j = 0; j < 8; ++j) sacc[j] = 0.0f;
#pragma unroll
            for (int kk = 0; kk < 2; ++kk) {
                v16bf kf = load_b_bf16(
                    Kb + ((size_t)(b * S_) + t0 + tt * 16) * D_ + h * HD_ + kk * 32, D_, lane);
                sacc = wmma_bf16(qf[kk], kf, sacc);
            }
            int tcol = t0 + tt * 16 + (lane & 15);
#pragma unroll
            for (int j = 0; j < 8; ++j) {
                int qloc = qt * 16 + j + ((lane >> 4) << 3);
                unsigned word = mbits[(size_t)(q0 + qloc) * (S_ / 32) + (tcol >> 5)];
                float v = ((word >> (tcol & 31)) & 1u) ? -1.0e9f : sacc[j] * 0.125f;
                Sbuf[qloc * 132 + tt * 16 + (lane & 15)] = v;
            }
        }
        __syncthreads();

        // ---- online softmax bookkeeping (4 threads per row)
        {
            int rr = tid >> 2, seg = tid & 3;
            float lmax = -3.0e38f;
            for (int c = seg * 32; c < seg * 32 + 32; ++c)
                lmax = fmaxf(lmax, Sbuf[rr * 132 + c]);
            red[rr * 4 + seg] = lmax;
        }
        __syncthreads();
        if ((tid & 3) == 0) {
            int rr = tid >> 2;
            float mc = fmaxf(fmaxf(red[rr*4], red[rr*4+1]), fmaxf(red[rr*4+2], red[rr*4+3]));
            float mold = m_s[rr];
            float mnew = fmaxf(mold, mc);
            a_s[rr] = __expf(mold - mnew);
            m_s[rr] = mnew;
        }
        __syncthreads();
        {
            int rr = tid >> 2, seg = tid & 3;
            float mnew = m_s[rr], ls = 0.f;
            for (int c = seg * 32; c < seg * 32 + 32; ++c) {
                float pv = __expf(Sbuf[rr * 132 + c] - mnew);
                Sbuf[rr * 132 + c] = pv;
                ls += pv;
            }
            red[rr * 4 + seg] = ls;
        }
        __syncthreads();
        if ((tid & 3) == 0) {
            int rr = tid >> 2;
            l_s[rr] = l_s[rr] * a_s[rr] + red[rr*4] + red[rr*4+1] + red[rr*4+2] + red[rr*4+3];
        }
        __syncthreads();

        // ---- rescale O accumulators by alpha, then O += P * V_chunk
#pragma unroll
        for (int e = 0; e < 2; ++e)
#pragma unroll
            for (int j = 0; j < 8; ++j)
                oacc[e][j] *= a_s[qt * 16 + j + ((lane >> 4) << 3)];

#pragma unroll
        for (int kk = 0; kk < 128; kk += 32) {
            v16bf pf = load_a_f32(&Sbuf[(qt * 16) * 132 + kk], 132, lane);
#pragma unroll
            for (int e = 0; e < 2; ++e) {
                v16bf vf = load_b_bf16(
                    Vt + ((size_t)(bh * HD_) + hh * 32 + e * 16) * S_ + t0 + kk, S_, lane);
                oacc[e] = wmma_bf16(pf, vf, oacc[e]);
            }
        }
        __syncthreads();
    }

    // ---- finalize: O /= l, write bf16
#pragma unroll
    for (int e = 0; e < 2; ++e)
#pragma unroll
        for (int j = 0; j < 8; ++j) {
            int qloc = qt * 16 + j + ((lane >> 4) << 3);
            float v = oacc[e][j] / l_s[qloc];
            int col = h * HD_ + hh * 32 + e * 16 + (lane & 15);
            Ob[((size_t)(b * S_) + q0 + qloc) * D_ + col] = f2bf(v);
        }
}

// ---------------- residual + layernorm-on-branch ----------------
__global__ __launch_bounds__(256) void k_ln_residual(
    const float* __restrict__ x, const float* __restrict__ y,
    const float* __restrict__ g, const float* __restrict__ beta,
    float* __restrict__ outF, bf16_t* __restrict__ outB)
{
    __shared__ float rs[256], rs2[256];
    const int row = blockIdx.x, tid = threadIdx.x;
    const float* yr = y + (size_t)row * D_;
    float s = 0.f, s2 = 0.f;
    for (int c = tid; c < D_; c += 256) { float v = yr[c]; s += v; s2 += v * v; }
    rs[tid] = s; rs2[tid] = s2;
    __syncthreads();
    for (int st = 128; st > 0; st >>= 1) {
        if (tid < st) { rs[tid] += rs[tid + st]; rs2[tid] += rs2[tid + st]; }
        __syncthreads();
    }
    float mean = rs[0] * (1.0f / D_);
    float var  = rs2[0] * (1.0f / D_) - mean * mean;
    float rstd = rsqrtf(var + EPSF);
    for (int c = tid; c < D_; c += 256) {
        size_t idx = (size_t)row * D_ + c;
        float v = x[idx] + (yr[c] - mean) * rstd * g[c] + beta[c];
        outF[idx] = v;
        if (outB) outB[idx] = f2bf(v);
    }
}

// ---------------- host orchestration ----------------
extern "C" void kernel_launch(void* const* d_in, const int* in_sizes, int n_in,
                              void* d_out, int out_size, void* d_ws, size_t ws_size,
                              hipStream_t stream)
{
    (void)in_sizes; (void)n_in; (void)out_size; (void)ws_size;
    const float* x    = (const float*)d_in[0];
    const int*   mask = (const int*)  d_in[1];
    const float* Wq   = (const float*)d_in[2];
    const float* Wk   = (const float*)d_in[3];
    const float* Wv   = (const float*)d_in[4];
    const float* Wo   = (const float*)d_in[5];
    const float* bo   = (const float*)d_in[6];
    const float* g1   = (const float*)d_in[7];
    const float* be1  = (const float*)d_in[8];
    const float* W1   = (const float*)d_in[9];
    const float* b1   = (const float*)d_in[10];
    const float* W2   = (const float*)d_in[11];
    const float* b2   = (const float*)d_in[12];
    const float* g2   = (const float*)d_in[13];
    const float* be2  = (const float*)d_in[14];
    float* out = (float*)d_out;

    char* p = (char*)d_ws;
    auto alloc = [&](size_t bytes) -> void* {
        void* r = (void*)p; p += (bytes + 255) & ~(size_t)255; return r;
    };
    bf16_t*  xb    = (bf16_t*)alloc((size_t)M_ * D_ * 2);
    bf16_t*  WqT   = (bf16_t*)alloc((size_t)D_ * D_ * 2);
    bf16_t*  WkT   = (bf16_t*)alloc((size_t)D_ * D_ * 2);
    bf16_t*  WvT   = (bf16_t*)alloc((size_t)D_ * D_ * 2);
    bf16_t*  WoT   = (bf16_t*)alloc((size_t)D_ * D_ * 2);
    bf16_t*  W1T   = (bf16_t*)alloc((size_t)DFF_ * D_ * 2);
    bf16_t*  W2T   = (bf16_t*)alloc((size_t)D_ * DFF_ * 2);
    bf16_t*  Qb    = (bf16_t*)alloc((size_t)M_ * D_ * 2);
    bf16_t*  Kb    = (bf16_t*)alloc((size_t)M_ * D_ * 2);
    bf16_t*  Vb    = (bf16_t*)alloc((size_t)M_ * D_ * 2);
    bf16_t*  Vt    = (bf16_t*)alloc((size_t)M_ * D_ * 2);
    unsigned* mbit = (unsigned*)alloc((size_t)S_ * (S_ / 32) * 4);
    bf16_t*  Ob    = (bf16_t*)alloc((size_t)M_ * D_ * 2);
    float*   attnF = (float*)alloc((size_t)M_ * D_ * 4);
    float*   x1F   = (float*)alloc((size_t)M_ * D_ * 4);
    bf16_t*  x1B   = (bf16_t*)alloc((size_t)M_ * D_ * 2);
    bf16_t*  h1B   = (bf16_t*)alloc((size_t)M_ * DFF_ * 2);
    float*   h2F   = (float*)alloc((size_t)M_ * D_ * 4);

    const dim3 tb(32, 8);
    int nx = M_ * D_;
    k_cvt_bf16<<<(nx + 255) / 256, 256, 0, stream>>>(x, xb, nx);
    k_transpose_cvt<<<dim3(D_/32, D_/32), tb, 0, stream>>>(Wq, WqT, D_, D_);
    k_transpose_cvt<<<dim3(D_/32, D_/32), tb, 0, stream>>>(Wk, WkT, D_, D_);
    k_transpose_cvt<<<dim3(D_/32, D_/32), tb, 0, stream>>>(Wv, WvT, D_, D_);
    k_transpose_cvt<<<dim3(D_/32, D_/32), tb, 0, stream>>>(Wo, WoT, D_, D_);
    k_transpose_cvt<<<dim3(DFF_/32, D_/32), tb, 0, stream>>>(W1, W1T, D_, DFF_);
    k_transpose_cvt<<<dim3(D_/32, DFF_/32), tb, 0, stream>>>(W2, W2T, DFF_, D_);
    k_maskbits<<<S_, 64, 0, stream>>>(mask, mbit);

    // Q/K/V projections (bf16 out)
    k_gemm<<<dim3(D_/128, M_/128), 256, 0, stream>>>(xb, WqT, nullptr, nullptr, Qb, M_, D_, D_, 0);
    k_gemm<<<dim3(D_/128, M_/128), 256, 0, stream>>>(xb, WkT, nullptr, nullptr, Kb, M_, D_, D_, 0);
    k_gemm<<<dim3(D_/128, M_/128), 256, 0, stream>>>(xb, WvT, nullptr, nullptr, Vb, M_, D_, D_, 0);
    k_transpose_v<<<dim3(S_/32, HD_/32, B_*H_), tb, 0, stream>>>(Vb, Vt);

    // attention
    k_attn<<<dim3(S_/64, B_*H_), 256, 0, stream>>>(Qb, Kb, Vt, mbit, Ob);

    // output projection + residual-with-norm-on-branch
    k_gemm<<<dim3(D_/128, M_/128), 256, 0, stream>>>(Ob, WoT, bo, attnF, nullptr, M_, D_, D_, 0);
    k_ln_residual<<<M_, 256, 0, stream>>>(x, attnF, g1, be1, x1F, x1B);

    // FFN with gelu after both matmuls
    k_gemm<<<dim3(DFF_/128, M_/128), 256, 0, stream>>>(x1B, W1T, b1, nullptr, h1B, M_, DFF_, D_, 1);
    k_gemm<<<dim3(D_/128, M_/128), 256, 0, stream>>>(h1B, W2T, b2, h2F, nullptr, M_, D_, DFF_, 1);
    k_ln_residual<<<M_, 256, 0, stream>>>(x1F, h2F, g2, be2, out, nullptr);
}
